// ParallelDeltaNet_70205535420763
// MI455X (gfx1250) — matrix operand order repaired and checked by reading, not compiled, not run on hardware
//
#include <hip/hip_runtime.h>
#include <hip/hip_fp16.h>

typedef _Float16 f16;
typedef __attribute__((ext_vector_type(16))) _Float16 v16h;
typedef __attribute__((ext_vector_type(8)))  _Float16 v8h;
typedef __attribute__((ext_vector_type(8)))  float    v8f;

#define Bb 2
#define Ll 4096
#define Dd 1024
#define Hh 16
#define DKk 64
#define MR (Bb*Ll)   // 8192 rows

__device__ __forceinline__ float sigm(float x){ return 1.0f/(1.0f+expf(-x)); }

// ---------------------------------------------------------------- WMMA helpers
__device__ __forceinline__ v8f wmma16(v16h a, v16h b, v8f c){
  // D = A(16x32 f16) * B(32x16 f16) + C(16x16 f32)
  return __builtin_amdgcn_wmma_f32_16x16x32_f16(false, a, false, b, (short)0, c,
                                                false, false);
}

__device__ __forceinline__ v16h cat8(v8h lo, v8h hi){
  return __builtin_shufflevector(lo, hi, 0,1,2,3,4,5,6,7,8,9,10,11,12,13,14,15);
}

// A fragment: 16x32 f16 from row-major [M,K] (ld = K stride).
// ISA layout: row = lane&15; lanes0-15 hold K = {0..7,16..23}, lanes16-31
// K = {8..15,24..31}. Per lane: two contiguous 8xf16 (16B) chunks.
__device__ __forceinline__ v16h load_a_frag(const f16* base, int ld){
  int lane = threadIdx.x & 31;
  int row = lane & 15, half = (lane >> 4) & 1;
  const f16* r = base + (size_t)row * ld + half * 8;
  v8h lo = *(const v8h*)(r);        // a[j]   = K = half*8 + j
  v8h hi = *(const v8h*)(r + 16);   // a[8+j] = K = 16 + half*8 + j
  return cat8(lo, hi);
}

// B fragment from TRANSPOSED source BT[N,K] row-major (ld = K stride).
// WMMA B (table order, matching A): b[j] = BT[col][half*8 + j], b[8+j] = BT[col][16 + half*8 + j]:
// per lane two contiguous 8xf16 (16B) chunks.
__device__ __forceinline__ v16h load_bT_frag(const f16* baseT, int ld){
  int lane = threadIdx.x & 31;
  int col = lane & 15, half = (lane >> 4) & 1;
  const f16* r = baseT + (size_t)col * ld + half * 8;
  v8h lo = *(const v8h*)(r);
  v8h hi = *(const v8h*)(r + 16);
  return cat8(lo, hi);
}

// ---------------------------------------------------------------- 0a: fp32->f16
__global__ __launch_bounds__(256)
void to_f16(const float* __restrict__ src, f16* __restrict__ dst, size_t n){
  size_t i = (size_t)blockIdx.x * blockDim.x + threadIdx.x;
  if (i < n) dst[i] = (f16)src[i];
}

// ------------------------------------- 0b: fp32 [R,C] -> f16 transposed [C,R]
__global__ __launch_bounds__(256)
void transpose_to_f16(const float* __restrict__ src, f16* __restrict__ dst,
                      int R, int C){
  __shared__ float tile[32][33];
  int c0 = blockIdx.x * 32, r0 = blockIdx.y * 32;
  int tx = threadIdx.x & 31, ty = threadIdx.x >> 5;   // 32x8
  for (int j = ty; j < 32; j += 8)
    tile[j][tx] = src[(size_t)(r0 + j) * C + c0 + tx];
  __syncthreads();
  for (int j = ty; j < 32; j += 8)
    dst[(size_t)(c0 + j) * R + r0 + tx] = (f16)tile[tx][j];
}

// --------------------------- 1: big GEMM  C[M,N] = A[M,K] @ BT[N,K]^T  (WMMA)
// block = 128 threads (4 waves), block tile 128x128, wave tile 64x64.
template<bool F32OUT>
__global__ __launch_bounds__(128)
void gemm_wmma(const f16* __restrict__ A, const f16* __restrict__ BT,
               void* __restrict__ Cp, int M, int N, int K)
{
  int wave = threadIdx.x >> 5;
  int wr = wave >> 1, wc = wave & 1;
  int m0 = blockIdx.y * 128 + wr * 64;
  int n0 = blockIdx.x * 128 + wc * 64;
  (void)M;
  v8f acc[4][4];
  #pragma unroll
  for (int i = 0; i < 4; i++)
    #pragma unroll
    for (int j = 0; j < 4; j++)
      acc[i][j] = (v8f){0.f,0.f,0.f,0.f,0.f,0.f,0.f,0.f};

  for (int k0 = 0; k0 < K; k0 += 32){
    if (k0 + 32 < K){
      __builtin_prefetch(A  + (size_t)m0 * K + k0 + 32, 0, 0);
      __builtin_prefetch(BT + (size_t)n0 * K + k0 + 32, 0, 0);
    }
    v16h bf[4];
    #pragma unroll
    for (int ni = 0; ni < 4; ni++)
      bf[ni] = load_bT_frag(BT + (size_t)(n0 + ni*16) * K + k0, K);
    #pragma unroll
    for (int mi = 0; mi < 4; mi++){
      v16h a = load_a_frag(A + (size_t)(m0 + mi*16) * K + k0, K);
      #pragma unroll
      for (int ni = 0; ni < 4; ni++)
        acc[mi][ni] = wmma16(a, bf[ni], acc[mi][ni]);
    }
  }

  int lane = threadIdx.x & 31;
  int col = lane & 15, half = (lane >> 4) & 1;
  #pragma unroll
  for (int mi = 0; mi < 4; mi++)
    #pragma unroll
    for (int ni = 0; ni < 4; ni++)
      #pragma unroll
      for (int i = 0; i < 8; i++){
        int r = m0 + mi*16 + half*8 + i;
        int c = n0 + ni*16 + col;
        if (F32OUT) ((float*)Cp)[(size_t)r * N + c] = acc[mi][ni][i];
        else        ((f16*)  Cp)[(size_t)r * N + c] = (f16)acc[mi][ni][i];
      }
}

// ---------------------------------------- 2: small projections x@{Wb,Wfd,Wsd}
__global__ __launch_bounds__(128)
void small_proj(const float* __restrict__ x,
                const float* __restrict__ Wb,
                const float* __restrict__ Wfd,
                const float* __restrict__ Wsd,
                float* __restrict__ sb, float* __restrict__ sfd,
                float* __restrict__ ssd)
{
  int row = blockIdx.x;
  __shared__ float xs[Dd];
  for (int i = threadIdx.x; i < Dd; i += blockDim.x)
    xs[i] = x[(size_t)row * Dd + i];
  __syncthreads();
  int t = threadIdx.x;
  if (t < 48){
    int p = t >> 4, c = t & 15;
    const float* W = (p == 0) ? Wb : (p == 1 ? Wfd : Wsd);
    float s = 0.f;
    for (int k = 0; k < Dd; k++) s += xs[k] * W[k * Hh + c];
    float* out = (p == 0) ? sb : (p == 1 ? sfd : ssd);
    out[(size_t)row * Hh + c] = s;   // pre-sigmoid, pre-bias
  }
}

// ----- 3: per-token/head: l2norm q,k; beta scaling; vT write; token-flux MLP
// block = 512 threads = 16 waves, one wave per head; grid = B*L rows.
__global__ __launch_bounds__(512)
void prep(f16* __restrict__ Qh, f16* __restrict__ Kh, const f16* __restrict__ Vh,
          f16* __restrict__ vT,              // [bh, 64, L] transposed v_scaled
          const float* __restrict__ sb,
          const float* __restrict__ Wtf1, const float* __restrict__ btf1,
          const float* __restrict__ Wtf2, const float* __restrict__ btf2,
          float* __restrict__ alpha)
{
  int row  = blockIdx.x;            // b*L + l
  int h    = threadIdx.x >> 5;      // head = wave id
  int lane = threadIdx.x & 31;
  int b = row / Ll, l = row % Ll;
  int bh = b * Hh + h;
  size_t base = (size_t)row * Dd + h * DKk;

  float q0 = (float)Qh[base + lane], q1 = (float)Qh[base + lane + 32];
  float k0 = (float)Kh[base + lane], k1 = (float)Kh[base + lane + 32];
  float v0 = (float)Vh[base + lane], v1 = (float)Vh[base + lane + 32];

  float qs = q0*q0 + q1*q1;
  float ks = k0*k0 + k1*k1;
  #pragma unroll
  for (int off = 16; off > 0; off >>= 1){
    qs += __shfl_xor(qs, off, 32);
    ks += __shfl_xor(ks, off, 32);
  }
  float qinv = 1.0f / fmaxf(sqrtf(qs), 1e-12f);
  float kinv = 1.0f / fmaxf(sqrtf(ks), 1e-12f);
  float beta = sigm(sb[(size_t)row * Hh + h]);

  float kb0 = k0 * kinv * beta, kb1 = k1 * kinv * beta;
  Qh[base + lane]      = (f16)(q0 * qinv);
  Qh[base + lane + 32] = (f16)(q1 * qinv);
  Kh[base + lane]      = (f16)kb0;
  Kh[base + lane + 32] = (f16)kb1;
  vT[((size_t)bh * DKk + lane)      * Ll + l] = (f16)(v0 * beta);
  vT[((size_t)bh * DKk + lane + 32) * Ll + l] = (f16)(v1 * beta);

  __shared__ float skb[Hh][DKk];
  skb[h][lane]      = kb0;
  skb[h][lane + 32] = kb1;
  __syncthreads();

  // token flux MLP: 64 -> 32 (silu) -> 1 (sigmoid, clip)
  float hm = btf1[lane];
  for (int kk = 0; kk < DKk; kk++) hm += skb[h][kk] * Wtf1[kk * 32 + lane];
  hm = hm * sigm(hm);                       // SiLU
  float t = hm * Wtf2[lane];
  #pragma unroll
  for (int off = 16; off > 0; off >>= 1) t += __shfl_xor(t, off, 32);
  if (lane == 0){
    float tf = sigm(t + btf2[0]);
    tf = fminf(fmaxf(tf, 0.01f), 0.99f);
    alpha[(size_t)bh * Ll + l] = 0.5f + 0.3f * tf;
  }
}

// ------ 4: per-(b,h) log-decay cumsum scan, decay weights, transposed f16 keys
__global__ __launch_bounds__(256)
void scan_decay(const float* __restrict__ sfd, const float* __restrict__ ssd,
                const float* __restrict__ bfd, const float* __restrict__ bsd,
                const f16* __restrict__ Kh,      // k_beta [b,l,h,dk]
                f16* __restrict__ kfT, f16* __restrict__ ksT, // [bh,dk,L]
                float* __restrict__ wsum_f, float* __restrict__ wsum_s,
                float* __restrict__ cs_buf)       // [bh,L]
{
  int bh = blockIdx.x;
  int b = bh / Hh, h = bh % Hh;
  int t = threadIdx.x;                      // 0..255
  __shared__ float sscan[256];
  float* cs = cs_buf + (size_t)bh * Ll;

  for (int which = 0; which < 2; which++){
    const float* sp = which ? ssd : sfd;
    float bias = which ? bsd[h] : bfd[h];

    // pass 1: inclusive cumsum of log(sigmoid(.)+1e-6) over L
    float run = 0.f;
    for (int c0 = 0; c0 < Ll; c0 += 256){
      int l = c0 + t;
      float xr = sp[((size_t)b * Ll + l) * Hh + h] + bias;
      float val = logf(sigm(xr) + 1e-6f);
      sscan[t] = val;
      __syncthreads();
      for (int off = 1; off < 256; off <<= 1){
        float add = (t >= off) ? sscan[t - off] : 0.f;
        __syncthreads();
        sscan[t] += add;
        __syncthreads();
      }
      float inc = sscan[t];
      float tot = sscan[255];
      __syncthreads();
      cs[l] = inc + run;
      run += tot;
    }
    float last = run;                       // cs[L-1]

    // pass 2: w = exp(last - cs); accumulate sum; write transposed f16 keys
    float wsloc = 0.f;
    f16* outT = (which ? ksT : kfT) + (size_t)bh * DKk * Ll;
    for (int c0 = 0; c0 < Ll; c0 += 256){
      int l = c0 + t;
      float wv = expf(last - cs[l]);        // exponent <= 0
      wsloc += wv;
      const f16* kbrow = Kh + ((size_t)b * Ll + l) * Dd + h * DKk;
      for (int e = 0; e < DKk; e++)
        outT[(size_t)e * Ll + l] = (f16)((float)kbrow[e] * wv);
    }
    sscan[t] = wsloc;
    __syncthreads();
    for (int off = 128; off > 0; off >>= 1){
      if (t < off) sscan[t] += sscan[t + off];
      __syncthreads();
    }
    if (t == 0) (which ? wsum_s : wsum_f)[bh] = sscan[0];
    __syncthreads();
  }
}

// ------- 5: state GEMM  M[64,64] = k_wT[64,L] @ vT[64,L]^T, /(wsum+1e-6)
// grid (B*H, 2); block 128 (4 waves split K); LDS ds_add_f32 reduction.
// Output is written TRANSPOSED (MT[n,d]) so the mix GEMM's B loads are contiguous.
__global__ __launch_bounds__(128)
void state_gemm(const f16* __restrict__ kfT, const f16* __restrict__ ksT,
                const f16* __restrict__ vT,
                const float* __restrict__ wsf, const float* __restrict__ wss,
                f16* __restrict__ MfT, f16* __restrict__ MsT)
{
  int bh = blockIdx.x;
  int which = blockIdx.y;
  const f16* Am = (which == 0 ? kfT : ksT) + (size_t)bh * DKk * Ll; // A [64,L]
  const f16* Bt = vT + (size_t)bh * DKk * Ll;                       // BT [64,L]
  int wave = threadIdx.x >> 5;

  v8f acc[4][4];
  #pragma unroll
  for (int i = 0; i < 4; i++)
    #pragma unroll
    for (int j = 0; j < 4; j++)
      acc[i][j] = (v8f){0.f,0.f,0.f,0.f,0.f,0.f,0.f,0.f};

  int kbeg = wave * (Ll / 4);
  for (int k0 = kbeg; k0 < kbeg + Ll / 4; k0 += 32){
    v16h bf[4];
    #pragma unroll
    for (int ni = 0; ni < 4; ni++)
      bf[ni] = load_bT_frag(Bt + (size_t)(ni*16) * Ll + k0, Ll);
    #pragma unroll
    for (int mi = 0; mi < 4; mi++){
      v16h a = load_a_frag(Am + (size_t)(mi*16) * Ll + k0, Ll);
      #pragma unroll
      for (int ni = 0; ni < 4; ni++)
        acc[mi][ni] = wmma16(a, bf[ni], acc[mi][ni]);
    }
  }

  __shared__ float sC[DKk * DKk];
  for (int i = threadIdx.x; i < DKk * DKk; i += blockDim.x) sC[i] = 0.f;
  __syncthreads();
  int lane = threadIdx.x & 31;
  int col = lane & 15, half = (lane >> 4) & 1;
  #pragma unroll
  for (int mi = 0; mi < 4; mi++)
    #pragma unroll
    for (int ni = 0; ni < 4; ni++)
      #pragma unroll
      for (int i = 0; i < 8; i++)
        atomicAdd(&sC[(mi*16 + half*8 + i) * DKk + ni*16 + col], acc[mi][ni][i]);
  __syncthreads();

  float ws = (which == 0 ? wsf[bh] : wss[bh]) + 1e-6f;
  f16* outp = (which == 0 ? MfT : MsT) + (size_t)bh * DKk * DKk;
  for (int i = threadIdx.x; i < DKk * DKk; i += blockDim.x){
    int d = i >> 6, n = i & 63;
    outp[(size_t)n * DKk + d] = (f16)(sC[i] / ws);   // transposed store
  }
}

// ------- 6: o = alpha*(q@Mf) + (1-alpha)*(q@Ms)  -> oh [b,l,h,dk] f16
// grid (L/256, B*H); block 128; both 64x64 B operands (transposed) in registers.
__global__ __launch_bounds__(128)
void mix_gemm(const f16* __restrict__ qn, const f16* __restrict__ MfT,
              const f16* __restrict__ MsT, const float* __restrict__ alpha,
              f16* __restrict__ oh)
{
  int bh = blockIdx.y;
  int b = bh / Hh, h = bh % Hh;
  int wave = threadIdx.x >> 5;
  int l0 = blockIdx.x * 256 + wave * 64;
  const f16* Ab = qn + (size_t)b * Ll * Dd + h * DKk;  // rows l, ld = Dd
  const f16* Bf = MfT + (size_t)bh * DKk * DKk;        // BT [N=64, K=64]
  const f16* Bs = MsT + (size_t)bh * DKk * DKk;

  v16h bff[2][4], bsf[2][4];
  #pragma unroll
  for (int ks = 0; ks < 2; ks++)
    #pragma unroll
    for (int ni = 0; ni < 4; ni++){
      bff[ks][ni] = load_bT_frag(Bf + (size_t)(ni*16) * DKk + ks*32, DKk);
      bsf[ks][ni] = load_bT_frag(Bs + (size_t)(ni*16) * DKk + ks*32, DKk);
    }

  int lane = threadIdx.x & 31;
  int col = lane & 15, half = (lane >> 4) & 1;
  for (int mi = 0; mi < 4; mi++){
    int lrow = l0 + mi * 16;
    v16h a0 = load_a_frag(Ab + (size_t)lrow * Dd,      Dd);
    v16h a1 = load_a_frag(Ab + (size_t)lrow * Dd + 32, Dd);
    float alv[8];
    #pragma unroll
    for (int i = 0; i < 8; i++)
      alv[i] = alpha[(size_t)bh * Ll + lrow + half*8 + i];
    #pragma unroll
    for (int ni = 0; ni < 4; ni++){
      v8f cf = (v8f){0.f,0.f,0.f,0.f,0.f,0.f,0.f,0.f};
      v8f cs = (v8f){0.f,0.f,0.f,0.f,0.f,0.f,0.f,0.f};
      cf = wmma16(a0, bff[0][ni], cf);
      cf = wmma16(a1, bff[1][ni], cf);
      cs = wmma16(a0, bsf[0][ni], cs);
      cs = wmma16(a1, bsf[1][ni], cs);
      #pragma unroll
      for (int i = 0; i < 8; i++){
        int l = lrow + half*8 + i;
        float o = alv[i] * cf[i] + (1.0f - alv[i]) * cs[i];
        oh[((size_t)b * Ll + l) * Dd + h * DKk + ni*16 + col] = (f16)o;
      }
    }
  }
}

// ---------------------------------------------------------------- host launcher
extern "C" void kernel_launch(void* const* d_in, const int* in_sizes, int n_in,
                              void* d_out, int out_size, void* d_ws, size_t ws_size,
                              hipStream_t stream)
{
  (void)in_sizes; (void)n_in; (void)out_size; (void)ws_size;
  const float* x    = (const float*)d_in[0];
  const float* Wq   = (const float*)d_in[1];
  const float* Wk   = (const float*)d_in[2];
  const float* Wv   = (const float*)d_in[3];
  const float* Wb   = (const float*)d_in[4];
  const float* Wo   = (const float*)d_in[5];
  const float* Wfd  = (const float*)d_in[6];
  const float* bfd  = (const float*)d_in[7];
  const float* Wsd  = (const float*)d_in[8];
  const float* bsd  = (const float*)d_in[9];
  const float* Wtf1 = (const float*)d_in[10];
  const float* btf1 = (const float*)d_in[11];
  const float* Wtf2 = (const float*)d_in[12];
  const float* btf2 = (const float*)d_in[13];
  float* out = (float*)d_out;

  char* wp = (char*)d_ws;
  auto take = [&](size_t bytes) -> char* {
    char* p = wp;
    wp += (bytes + 255) & ~(size_t)255;
    return p;
  };

  f16* xh    = (f16*)take((size_t)MR * Dd * 2);
  f16* WqT   = (f16*)take((size_t)Dd * Dd * 2);   // transposed [N,K] f16
  f16* WkT   = (f16*)take((size_t)Dd * Dd * 2);
  f16* WvT   = (f16*)take((size_t)Dd * Dd * 2);
  f16* WoT   = (f16*)take((size_t)Dd * Dd * 2);
  f16* Qh    = (f16*)take((size_t)MR * Dd * 2);
  f16* Kh    = (f16*)take((size_t)MR * Dd * 2);
  f16* Vh    = (f16*)take((size_t)MR * Dd * 2);
  f16* vT    = (f16*)take((size_t)Bb * Hh * DKk * Ll * 2);  // [bh,64,L]
  float* sb    = (float*)take((size_t)MR * Hh * 4);
  float* sfd   = (float*)take((size_t)MR * Hh * 4);
  float* ssd   = (float*)take((size_t)MR * Hh * 4);
  float* alpha = (float*)take((size_t)Bb * Hh * Ll * 4);
  float* csb   = (float*)take((size_t)Bb * Hh * Ll * 4);
  float* wsf   = (float*)take((size_t)Bb * Hh * 4);
  float* wss   = (float*)take((size_t)Bb * Hh * 4);
  f16* kfT   = (f16*)take((size_t)Bb * Hh * DKk * Ll * 2);
  f16* ksT   = (f16*)take((size_t)Bb * Hh * DKk * Ll * 2);
  f16* MfT   = (f16*)take((size_t)Bb * Hh * DKk * DKk * 2);
  f16* MsT   = (f16*)take((size_t)Bb * Hh * DKk * DKk * 2);
  f16* oh    = (f16*)take((size_t)MR * Dd * 2);

  // 0: convert activations; transpose+convert big weights
  {
    size_t n = (size_t)MR * Dd;
    to_f16<<<dim3((unsigned)((n + 255) / 256)), 256, 0, stream>>>(x, xh, n);
    dim3 gt(Dd / 32, Dd / 32);
    transpose_to_f16<<<gt, 256, 0, stream>>>(Wq, WqT, Dd, Dd);
    transpose_to_f16<<<gt, 256, 0, stream>>>(Wk, WkT, Dd, Dd);
    transpose_to_f16<<<gt, 256, 0, stream>>>(Wv, WvT, Dd, Dd);
    transpose_to_f16<<<gt, 256, 0, stream>>>(Wo, WoT, Dd, Dd);
  }

  // 1: Q/K/V projections (WMMA)
  dim3 gg(Dd / 128, MR / 128);   // (8, 64)
  gemm_wmma<false><<<gg, 128, 0, stream>>>(xh, WqT, Qh, MR, Dd, Dd);
  gemm_wmma<false><<<gg, 128, 0, stream>>>(xh, WkT, Kh, MR, Dd, Dd);
  gemm_wmma<false><<<gg, 128, 0, stream>>>(xh, WvT, Vh, MR, Dd, Dd);

  // 2: small projections
  small_proj<<<MR, 128, 0, stream>>>(x, Wb, Wfd, Wsd, sb, sfd, ssd);

  // 3: per-token/head prep (l2norm, beta, vT scatter, token-flux MLP, alpha)
  prep<<<MR, 512, 0, stream>>>(Qh, Kh, Vh, vT, sb, Wtf1, btf1, Wtf2, btf2, alpha);

  // 4: decay scans + transposed decay-weighted keys
  scan_decay<<<Bb * Hh, 256, 0, stream>>>(sfd, ssd, bfd, bsd, Kh, kfT, ksT,
                                          wsf, wss, csb);

  // 5: KV outer-product states (WMMA), transposed output
  state_gemm<<<dim3(Bb * Hh, 2), 128, 0, stream>>>(kfT, ksT, vT, wsf, wss, MfT, MsT);

  // 6: q@M blend (WMMA)
  mix_gemm<<<dim3(Ll / 256, Bb * Hh), 128, 0, stream>>>(Qh, MfT, MsT, alpha, oh);

  // 7: output projection (WMMA, f32 out)
  gemm_wmma<true><<<gg, 128, 0, stream>>>(oh, WoT, out, MR, Dd, Dd);
}
